// FFTBias_36438502540200
// MI455X (gfx1250) — matrix operand order repaired and checked by reading, not compiled, not run on hardware
//
#include <hip/hip_runtime.h>
#include <hip/hip_bf16.h>

#define S_LEN 2046
#define MAXP  2048
#define NH    12
#define NB    8
#define ND    64

#define BM 128          // rows (n) per block: 8 waves x 16
#define BN 128          // cols per block: 2 batches x 64 d
#define BK 32
#define BSTRIDE 40      // bf16 elems per Bs row (80B rows -> 16B aligned, bank-staggered)
#define EM_LEN 4096     // mirrored e: eM[j] = e[|j-2047|]

typedef __attribute__((ext_vector_type(16))) __bf16 v16bf;
typedef __attribute__((ext_vector_type(8)))  __bf16 v8bf;
typedef __attribute__((ext_vector_type(8)))  float  v8f;

// ---------------------------------------------------------------------------
// Kernel 1: z_pb[0, p, h]  (Toeplitz row sums via prefix-sum identity)
// rowsum[n] = P(n) + P(S-1-n) - e[0],  P = inclusive prefix of e
// ---------------------------------------------------------------------------
__global__ __launch_bounds__(256) void zpb_kernel(const float* __restrict__ w,
                                                  const float* __restrict__ off,
                                                  float* __restrict__ zout) {
  __shared__ float eF[MAXP];
  __shared__ float P[MAXP];
  __shared__ float csum[256];
  const int h = blockIdx.x;
  const int tid = threadIdx.x;
  const float o = off[h];
  for (int d = tid; d < MAXP; d += 256)
    eF[d] = (d < S_LEN) ? __expf(w[h * S_LEN + d] - o) : 0.0f;
  __syncthreads();
  float s = 0.f;
#pragma unroll
  for (int j = 0; j < 8; ++j) s += eF[tid * 8 + j];
  csum[tid] = s;
  __syncthreads();
  if (tid == 0) {
    float run = 0.f;
    for (int i = 0; i < 256; ++i) { float t = csum[i]; csum[i] = run; run += t; }
  }
  __syncthreads();
  float run = csum[tid];
#pragma unroll
  for (int j = 0; j < 8; ++j) { run += eF[tid * 8 + j]; P[tid * 8 + j] = run; }
  __syncthreads();
  const float e0 = eF[0];
  for (int n = tid; n < S_LEN; n += 256)
    zout[(1 + n) * NH + h] = P[n] + P[S_LEN - 1 - n] - e0;
  if (tid == 0) { zout[h] = 0.f; zout[(MAXP - 1) * NH + h] = 0.f; }
}

// ---------------------------------------------------------------------------
// Kernel 2: zero boundary rows pbv[:,0,:,:] and pbv[:,2047,:,:]
// ---------------------------------------------------------------------------
__global__ void edge_kernel(float* __restrict__ out) {
  int i = blockIdx.x * blockDim.x + threadIdx.x;
  const int tot = 2 * NB * NH * ND;
  if (i >= tot) return;
  int side = i / (NB * NH * ND);
  int rem  = i % (NB * NH * ND);
  int b    = rem / (NH * ND);
  int hd   = rem % (NH * ND);
  int p    = side ? (MAXP - 1) : 0;
  out[((size_t)(b * MAXP + p)) * (NH * ND) + hd] = 0.f;
}

// ---------------------------------------------------------------------------
// Kernel 3: one-pass convert+transpose  v(f32,[b,p,h,d]) -> vT(bf16,[b,h,d,t])
// t-contiguous rows of 2048 (zero-padded for t >= S so the GEMM needs no
// tail masks). 64x64 LDS tile, stride 65 to avoid bank conflicts.
// ---------------------------------------------------------------------------
__global__ __launch_bounds__(256) void transpose_kernel(
    const float* __restrict__ v, __bf16* __restrict__ vT) {
  __shared__ float tile[64][65];
  const int tid = threadIdx.x;
  const int t0  = blockIdx.x * 64;
  const int b   = blockIdx.y;
  const int h   = blockIdx.z;

  // coalesced read over d (zero for t >= S)
#pragma unroll
  for (int it = 0; it < 16; ++it) {
    const int e  = tid + it * 256;
    const int tt = e >> 6;
    const int d  = e & 63;
    const int t  = t0 + tt;
    tile[tt][d] = (t < S_LEN)
        ? v[(((size_t)b * MAXP + 1 + t) * NH + h) * ND + d] : 0.0f;
  }
  __syncthreads();

  // packed bf16 write along t: thread -> (d = tid/4, 16-wide t segment)
  const int d  = tid >> 2;
  const int ts = (tid & 3) * 16;
  v8bf o0, o1;
#pragma unroll
  for (int j = 0; j < 8; ++j) o0[j] = (__bf16)tile[ts + j][d];
#pragma unroll
  for (int j = 0; j < 8; ++j) o1[j] = (__bf16)tile[ts + 8 + j][d];
  __bf16* dst = vT + (((size_t)b * NH + h) * ND + d) * MAXP + t0 + ts;
  *(v8bf*)(dst)     = o0;
  *(v8bf*)(dst + 8) = o1;
}

// ---------------------------------------------------------------------------
// Kernel 4: per-head symmetric-Toeplitz GEMM via v_wmma_f32_16x16x32_bf16
//   P[n, c] = sum_t e[|n-t|] * v[b, 1+t, h, c]
// Mirror trick: eM[j] = e[|j-2047|]  =>  A[m][t] = eM[2047 + t - m]
//   -> per-lane A fragment = two ds_load_b128 (no abs, no masks).
// B tile: pure aligned copy vT -> LDS (bf16, already transposed & padded):
//   1x global b128 + 1x ds_store_b128 per thread per K-step, prefetched one
//   step ahead in registers to overlap HBM latency with the WMMA burst.
// 8 waves = 8 M-subtiles; each wave: 1 afrag x 8 bfrags -> 8 WMMAs into
// 8 independent accumulators (no D->A/B hazards, back-to-back issue).
// ---------------------------------------------------------------------------
__global__ __launch_bounds__(256) void toeplitz_mm_kernel(
    const __bf16* __restrict__ vT, const float* __restrict__ off,
    const float* __restrict__ w, float* __restrict__ out) {
  __shared__ __bf16 eM[EM_LEN];
  __shared__ __bf16 Bs[BN * BSTRIDE];

  const int tid   = threadIdx.x;
  const int n0    = blockIdx.x * BM;
  const int bPair = blockIdx.y;          // covers batches 2*bPair, 2*bPair+1
  const int h     = blockIdx.z;

  // eM[j] = exp(w[h,|j-2047|] - offset[h]) for |j-2047| < S, else 0
  {
    const float o = off[h];
    for (int j = tid; j < EM_LEN; j += 256) {
      int d = j - 2047; d = d < 0 ? -d : d;
      eM[j] = (__bf16)((d < S_LEN) ? __expf(w[h * S_LEN + d] - o) : 0.0f);
    }
  }

  const int wave   = tid >> 5;           // M subtile (0..7)
  const int lane   = tid & 31;
  const int lane15 = lane & 15;
  const int laneHi = lane >> 4;

  v8f acc[8] = {};

  // B staging: thread -> (row c = tid/2, 16-element half of the 32-k row)
  const int cRow = tid >> 1;
  const int half = tid & 1;
  const __bf16* srcRow = vT +
      (((size_t)(bPair * 2 + (cRow >> 6)) * NH + h) * ND + (cRow & 63)) * MAXP +
      half * 16;
  __bf16* dstRow = &Bs[cRow * BSTRIDE + half * 16];

  uint4 pre;                              // 8 bf16 prefetched per thread
  auto load_tile  = [&](int t0) { pre = *(const uint4*)(srcRow + t0); };
  auto store_tile = [&]()       { *(uint4*)dstRow = pre; };

  load_tile(0);
  __syncthreads();                        // eM ready

  const int m = n0 + wave * 16 + lane15;  // this lane's A row

  for (int t0 = 0; t0 < S_LEN; t0 += BK) {
    store_tile();
    __syncthreads();                      // Bs ready

    if (t0 + BK < S_LEN) load_tile(t0 + BK);   // overlap with WMMA burst

    // A fragment (contiguous k order, matched by B): lanes 0-15 K=0..15,
    // lanes 16-31 K=16..31; consecutive in eM via mirror trick.
    const int a0 = 2047 + t0 - m + laneHi * 16;
    v16bf afrag;
#pragma unroll
    for (int i = 0; i < 8; ++i) afrag[i]     = eM[a0 + i];
#pragma unroll
    for (int i = 0; i < 8; ++i) afrag[8 + i] = eM[a0 + 8 + i];

#pragma unroll
    for (int ns = 0; ns < 8; ++ns) {
      // B fragment (contiguous k order, matches A): lane<16 -> N=lane, K=0..15;
      // lane>=16 -> N=lane-16, K=16..31. Rows of Bs are 16B-aligned.
      const int cb = (ns * 16 + lane15) * BSTRIDE + laneHi * 16;
      v16bf bfrag;
#pragma unroll
      for (int i = 0; i < 16; ++i) bfrag[i] = Bs[cb + i];
      acc[ns] = __builtin_amdgcn_wmma_f32_16x16x32_bf16(
          false, afrag, false, bfrag, (short)0, acc[ns], false, false);
    }
    __syncthreads();                      // compute done before Bs overwrite
  }

  // Store D (32-bit C/D 16x16): lanes 0-15 -> (M=r, N=lane),
  // lanes 16-31 -> (M=8+r, N=lane-16)
#pragma unroll
  for (int ns = 0; ns < 8; ++ns) {
    const int c = ns * 16 + lane15;
    const int bOut = bPair * 2 + (c >> 6);
    const int d = c & 63;
#pragma unroll
    for (int r = 0; r < 8; ++r) {
      const int mm = n0 + wave * 16 + laneHi * 8 + r;
      if (mm < S_LEN)
        out[(((size_t)bOut * MAXP + 1 + mm) * NH + h) * ND + d] = acc[ns][r];
    }
  }
}

// ---------------------------------------------------------------------------
extern "C" void kernel_launch(void* const* d_in, const int* in_sizes, int n_in,
                              void* d_out, int out_size, void* d_ws, size_t ws_size,
                              hipStream_t stream) {
  (void)in_sizes; (void)n_in; (void)out_size; (void)ws_size;
  const float* v   = (const float*)d_in[0];
  const float* off = (const float*)d_in[1];
  const float* w   = (const float*)d_in[2];
  float*  out  = (float*)d_out;
  float*  zout = out + (size_t)NB * MAXP * NH * ND;   // z_pb after pbv
  __bf16* vT   = (__bf16*)d_ws;                       // 8*12*64*2048 bf16 = 25 MB

  zpb_kernel<<<NH, 256, 0, stream>>>(w, off, zout);

  const int tot = 2 * NB * NH * ND;
  edge_kernel<<<(tot + 255) / 256, 256, 0, stream>>>(out);

  dim3 tgrid(MAXP / 64, NB, NH);
  transpose_kernel<<<tgrid, 256, 0, stream>>>(v, vT);

  dim3 grid((S_LEN + BM - 1) / BM, NB / 2, NH);
  toeplitz_mm_kernel<<<grid, 256, 0, stream>>>(vT, off, w, out);
}
